// SelfAttention_89833535963953
// MI455X (gfx1250) — matrix operand order repaired and checked by reading, not compiled, not run on hardware
//
#include <hip/hip_runtime.h>
#include <hip/hip_bf16.h>

#define BATCH 4
#define CH    128
#define NPIX  4096
#define DQK   32

typedef __attribute__((ext_vector_type(16))) __bf16 v16bf;
typedef __attribute__((ext_vector_type(8)))  float  v8f;

union BF16Tile {
    v16bf    v;
    uint4    u4[2];
    unsigned u[8];
};

__device__ __forceinline__ unsigned pack_bf16(float a, float b) {
    union { __bf16 h[2]; unsigned u; } r;
    r.h[0] = (__bf16)a;
    r.h[1] = (__bf16)b;
    return r.u;
}

// xor-16 half-wave exchange in the VALU pipe (v_permlanex16_b32, identity
// selects) -- no LDS hardware, no DScnt waits.
__device__ __forceinline__ float xor16_f(float v) {
    int i = __float_as_int(v);
    return __int_as_float(__builtin_amdgcn_permlanex16(
        i, i, 0x76543210, 0xfedcba98, true, false));
}
__device__ __forceinline__ unsigned xor16_u(unsigned v) {
    return (unsigned)__builtin_amdgcn_permlanex16(
        (int)v, (int)v, 0x76543210, 0xfedcba98, true, false);
}

// ---------------------------------------------------------------------------
// Kernel 1: pe bilinear upsample + xp = x + 0.025*pe, then 1x1-conv
// projections (q,k scaled+bf16 [n][32]; v bf16 [c][n]); xp kept f32 for the
// residual. One block = 16 pixels, 256 threads.
// ---------------------------------------------------------------------------
__global__ void __launch_bounds__(256) proj_kernel(
    const float* __restrict__ x,  const float* __restrict__ Wq,
    const float* __restrict__ bq, const float* __restrict__ Wk,
    const float* __restrict__ bk, const float* __restrict__ Wv,
    const float* __restrict__ bv, const float* __restrict__ pos,
    float* __restrict__ xp, __bf16* __restrict__ qT,
    __bf16* __restrict__ kT, __bf16* __restrict__ vC) {
    __shared__ float xs[CH * 16];
    const int t  = threadIdx.x;
    const int b  = blockIdx.x >> 8;           // 256 pixel-tiles per batch
    const int n0 = (blockIdx.x & 255) << 4;
    const float* xb  = x  + (size_t)b * CH * NPIX;
    float*       xpb = xp + (size_t)b * CH * NPIX;
    // softmax scale 1/(sqrt(128)+1e-8) and log2(e) folded into q
    const float QSCALE = 1.4426950408889634f / (11.313708498984761f + 1e-8f);

#pragma unroll
    for (int i = 0; i < 8; ++i) {
        int e = t + i * 256;
        int c = e >> 4, pix = e & 15;
        int n = n0 + pix;
        int y = n >> 6, xxi = n & 63;
        float sy = 0.5f * (float)y  - 0.25f;
        float sx = 0.5f * (float)xxi - 0.25f;
        float fy = floorf(sy), fx = floorf(sx);
        float wy = sy - fy,    wx = sx - fx;
        int iy = (int)fy, ix = (int)fx;
        int y0 = iy < 0 ? 0 : iy;      int y1 = iy + 1 > 31 ? 31 : iy + 1;
        int x0 = ix < 0 ? 0 : ix;      int x1 = ix + 1 > 31 ? 31 : ix + 1;
        const float* pp = pos + c * 1024;
        float v00 = pp[y0 * 32 + x0], v01 = pp[y0 * 32 + x1];
        float v10 = pp[y1 * 32 + x0], v11 = pp[y1 * 32 + x1];
        float pe  = (1.f - wy) * ((1.f - wx) * v00 + wx * v01)
                  +        wy  * ((1.f - wx) * v10 + wx * v11);
        float val = xb[c * NPIX + n] + 0.025f * pe;
        xs[c * 16 + pix] = val;
        xpb[c * NPIX + n] = val;
    }
    __syncthreads();

    __bf16* qTb = qT + (size_t)b * NPIX * DQK;
    __bf16* kTb = kT + (size_t)b * NPIX * DQK;
    __bf16* vCb = vC + (size_t)b * CH * NPIX;
#pragma unroll
    for (int i = 0; i < 12; ++i) {
        int e = t + i * 256;
        int o = e >> 4, pix = e & 15;
        int n = n0 + pix;
        const float* wrow;
        float acc;
        if (o < 32)      { wrow = Wq + o * CH;        acc = bq[o]; }
        else if (o < 64) { wrow = Wk + (o - 32) * CH; acc = bk[o - 32]; }
        else             { wrow = Wv + (o - 64) * CH; acc = bv[o - 64]; }
        for (int c = 0; c < CH; ++c) acc += wrow[c] * xs[c * 16 + pix];
        if (o < 32)      qTb[n * DQK + o]                 = (__bf16)(acc * QSCALE);
        else if (o < 64) kTb[n * DQK + (o - 32)]          = (__bf16)acc;
        else             vCb[(size_t)(o - 64) * NPIX + n] = (__bf16)acc;
    }
}

// ---------------------------------------------------------------------------
// Online-softmax update for one 16-query set over a 32-key step.
// st0/st1 are S^T tiles (key rows in regs, query in lane). Produces the
// P^T B-operand and the accumulator rescale factor.
// ---------------------------------------------------------------------------
__device__ __forceinline__ float softmax_step(
    const v8f& st0, const v8f& st1, float& m_run, float& l_run,
    BF16Tile& bp, int hi) {
    float mt = st0[0];
#pragma unroll
    for (int r = 1; r < 8; ++r) mt = fmaxf(mt, st0[r]);
#pragma unroll
    for (int r = 0; r < 8; ++r) mt = fmaxf(mt, st1[r]);
    mt = fmaxf(mt, xor16_f(mt));
    float mnew  = fmaxf(m_run, mt);
    float alpha = __builtin_amdgcn_exp2f(m_run - mnew);   // raw v_exp_f32
    float p0[8], p1[8], ls = 0.f;
#pragma unroll
    for (int r = 0; r < 8; ++r) {
        p0[r] = __builtin_amdgcn_exp2f(st0[r] - mnew);
        p1[r] = __builtin_amdgcn_exp2f(st1[r] - mnew);
        ls += p0[r] + p1[r];
    }
    ls += xor16_f(ls);
    l_run = l_run * alpha + ls;
    m_run = mnew;
    unsigned pk0[4], pk1[4];
#pragma unroll
    for (int j = 0; j < 4; ++j) {
        pk0[j] = pack_bf16(p0[2 * j], p0[2 * j + 1]);
        pk1[j] = pack_bf16(p1[2 * j], p1[2 * j + 1]);
    }
#pragma unroll
    for (int j = 0; j < 4; ++j) {
        unsigned o0 = xor16_u(pk0[j]);
        unsigned o1 = xor16_u(pk1[j]);
        bp.u[j]     = hi ? o1     : pk0[j];  // keys kbase+0..7
        bp.u[4 + j] = hi ? pk1[j] : o0;      // keys kbase+8..15
    }
    return alpha;
}

// ---------------------------------------------------------------------------
// Kernel 2: flash attention with bf16 WMMA.
// Work split: each wave owns 32 queries (two 16-query B-operand tiles that
// share every K-tile and V-tile load -> 2x arithmetic intensity vs L2) but
// only HALF of the 128 output channels (wave-pair split) so the accumulator
// set is 64 VGPRs and the whole wave stays under the 256-VGPR direct-
// addressing limit (no s_set_vgpr_msb traffic). S^T = K_tile(A) x Q^T(B)
// puts the query in the lane: softmax reduction is in-register + one
// permlanex16. O^T += V^T(A) x P^T(B). No LDS, no barriers: operands are
// L2-resident (192MB L2 >> 6MB working set).
// ---------------------------------------------------------------------------
__global__ void __launch_bounds__(256) attn_kernel(
    const __bf16* __restrict__ qT, const __bf16* __restrict__ kT,
    const __bf16* __restrict__ vC, const float* __restrict__ xp,
    const float* __restrict__ gammap, float* __restrict__ out) {
    const int b      = blockIdx.x >> 5;          // 32 blocks per batch
    const int qblock = (blockIdx.x & 31) << 7;   // 128 queries per block
    const int wave   = threadIdx.x >> 5;
    const int lane   = threadIdx.x & 31;
    const int q      = lane & 15;                // query column (B/C/D lane)
    const int hi     = lane >> 4;                // half-wave select
    const int kb     = hi << 4;                  // A-operand K base (16*hi)
    const int q0     = qblock + ((wave >> 1) << 5); // 32 queries per wave
    const int ch0    = (wave & 1) << 6;          // channel half: 0 or 64
    const int nqA    = q0 + q;
    const int nqB    = q0 + 16 + q;

    const __bf16* qTb = qT + (size_t)b * NPIX * DQK;
    const __bf16* kTb = kT + (size_t)b * NPIX * DQK;
    const __bf16* vCb = vC + (size_t)b * CH * NPIX;
    const float*  xpb = xp + (size_t)b * CH * NPIX;
    float*       outb = out + (size_t)b * CH * NPIX;

    // Two Q B-operands: lane q holds 16 contiguous bf16 at d-base hi*16
    BF16Tile qbA, qbB;
    {
        const __bf16* qpA = qTb + nqA * DQK + (hi << 4);
        qbA.u4[0] = *(const uint4*)(qpA);
        qbA.u4[1] = *(const uint4*)(qpA + 8);
        const __bf16* qpB = qTb + nqB * DQK + (hi << 4);
        qbB.u4[0] = *(const uint4*)(qpB);
        qbB.u4[1] = *(const uint4*)(qpB + 8);
    }

    v8f accA[4], accB[4];
    v8f zero = {};
#pragma unroll
    for (int cc = 0; cc < 4; ++cc) { accA[cc] = zero; accB[cc] = zero; }
    float mA = -1e30f, lA = 0.f, mB = -1e30f, lB = 0.f;

    for (int j0 = 0; j0 < NPIX; j0 += 32) {
        // K' A-tiles (shared by both query sets)
        const __bf16* k0p = kTb + (size_t)(j0 + q) * DQK;
        const __bf16* k1p = kTb + (size_t)(j0 + 16 + q) * DQK;
        BF16Tile ka0, ka1;
        ka0.u4[0] = *(const uint4*)(k0p + kb);
        ka0.u4[1] = *(const uint4*)(k0p + kb + 8);
        ka1.u4[0] = *(const uint4*)(k1p + kb);
        ka1.u4[1] = *(const uint4*)(k1p + kb + 8);

        v8f stA0 = __builtin_amdgcn_wmma_f32_16x16x32_bf16(
            false, ka0.v, false, qbA.v, (short)0, zero, false, false);
        v8f stA1 = __builtin_amdgcn_wmma_f32_16x16x32_bf16(
            false, ka1.v, false, qbA.v, (short)0, zero, false, false);
        v8f stB0 = __builtin_amdgcn_wmma_f32_16x16x32_bf16(
            false, ka0.v, false, qbB.v, (short)0, zero, false, false);
        v8f stB1 = __builtin_amdgcn_wmma_f32_16x16x32_bf16(
            false, ka1.v, false, qbB.v, (short)0, zero, false, false);

        BF16Tile bpA, bpB;
        float alphaA = softmax_step(stA0, stA1, mA, lA, bpA, hi);
        float alphaB = softmax_step(stB0, stB1, mB, lB, bpB, hi);

#pragma unroll
        for (int cc = 0; cc < 4; ++cc)
#pragma unroll
            for (int r = 0; r < 8; ++r) {
                accA[cc][r] *= alphaA;
                accB[cc][r] *= alphaB;
            }

        // O^T += V^T_chunk(A) x P^T(B); V tile shared by both query sets
#pragma unroll
        for (int cc = 0; cc < 4; ++cc) {
            const __bf16* vp =
                vCb + (size_t)(ch0 + cc * 16 + q) * NPIX + j0 + kb;
            BF16Tile va;
            va.u4[0] = *(const uint4*)(vp);
            va.u4[1] = *(const uint4*)(vp + 8);
            accA[cc] = __builtin_amdgcn_wmma_f32_16x16x32_bf16(
                false, va.v, false, bpA.v, (short)0, accA[cc], false, false);
            accB[cc] = __builtin_amdgcn_wmma_f32_16x16x32_bf16(
                false, va.v, false, bpB.v, (short)0, accB[cc], false, false);
        }
    }

    // epilogue: out = gamma * O / l + xp
    float invlA = 1.0f / lA;
    float invlB = 1.0f / lB;
    float g     = gammap[0];
#pragma unroll
    for (int cc = 0; cc < 4; ++cc)
#pragma unroll
        for (int r = 0; r < 8; ++r) {
            int c = ch0 + cc * 16 + r + (hi << 3);
            size_t iA = (size_t)c * NPIX + nqA;
            size_t iB = (size_t)c * NPIX + nqB;
            outb[iA] = g * accA[cc][r] * invlA + xpb[iA];
            outb[iB] = g * accB[cc][r] * invlB + xpb[iB];
        }
}

// ---------------------------------------------------------------------------
extern "C" void kernel_launch(void* const* d_in, const int* in_sizes, int n_in,
                              void* d_out, int out_size, void* d_ws, size_t ws_size,
                              hipStream_t stream) {
    const float* x     = (const float*)d_in[0];
    const float* Wq    = (const float*)d_in[1];
    const float* bq    = (const float*)d_in[2];
    const float* Wk    = (const float*)d_in[3];
    const float* bk    = (const float*)d_in[4];
    const float* Wv    = (const float*)d_in[5];
    const float* bv    = (const float*)d_in[6];
    const float* gamma = (const float*)d_in[7];
    const float* pos   = (const float*)d_in[8];
    float* out = (float*)d_out;

    char* ws = (char*)d_ws;
    float*  xp = (float*)ws;   ws += (size_t)BATCH * CH * NPIX * sizeof(float);
    __bf16* qT = (__bf16*)ws;  ws += (size_t)BATCH * NPIX * DQK * sizeof(__bf16);
    __bf16* kT = (__bf16*)ws;  ws += (size_t)BATCH * NPIX * DQK * sizeof(__bf16);
    __bf16* vC = (__bf16*)ws;

    proj_kernel<<<dim3(BATCH * NPIX / 16), dim3(256), 0, stream>>>(
        x, Wq, bq, Wk, bk, Wv, bv, pos, xp, qT, kT, vC);
    attn_kernel<<<dim3(BATCH * NPIX / 128), dim3(256), 0, stream>>>(
        qT, kT, vC, xp, gamma, out);
}
